// SelfAttention3D_23605140258916
// MI455X (gfx1250) — matrix operand order repaired and checked by reading, not compiled, not run on hardware
//
#include <hip/hip_runtime.h>
#include <hip/hip_fp16.h>
#include <math.h>

// Problem constants (fixed by the reference).
constexpr int B_ = 2;
constexpr int N_ = 4096;      // D*H*W = 16*16*16
constexpr int C_ = 64;
constexpr int CB_ = 32;       // CBAR
constexpr int QT_ = N_ / 16;  // 256 query tiles of 16
constexpr int KT32_ = N_ / 32; // 128 key tiles of 32

typedef __attribute__((ext_vector_type(16))) _Float16 v16h;
typedef __attribute__((ext_vector_type(8)))  float    v8f;

__device__ __forceinline__ v8f wmma16x16x32(v16h a, v16h b, v8f c) {
  // (neg_a, A, neg_b, B, c_mod, C, reuse_a, reuse_b)
  return __builtin_amdgcn_wmma_f32_16x16x32_f16(false, a, false, b, (short)0, c, false, false);
}

// Cross-half (lane ^ 16) exchange in wave32 via v_permlanex16_b32:
// pure VALU, no LDS round trip, no s_wait_dscnt serialization.
__device__ __forceinline__ float xor16(float v) {
  int i = __builtin_bit_cast(int, v);
  int r = __builtin_amdgcn_permlanex16(i, i, 0x76543210, 0xfedcba98, false, false);
  return __builtin_bit_cast(float, r);
}

// -----------------------------------------------------------------------------
// Kernel 1: per-voxel projections f,g,h = x @ W + b (fp32, exact), then split
// each value into f16 hi/lo and scatter into wave32 WMMA fragment layouts:
//   G (queries)  -> B-fragment of 32x16 (K=cbar, N=query):
//        lane = m + 16*((cb%16)>=8), slot = (cb&7) + 8*(cb>=16)
//   F (keys)     -> A-fragment of 16x32 (M=key, K=cbar):
//        lane = m + 16*((cb%16)>=8), slot = (cb&7) + 8*(cb>=16)
//   V=h (values) -> A-fragment of V^T, 16x32 (M=cb%16, K=key-in-tile), per cb half:
//        lane = (cb%16) + 16*(kk>=16), slot = kk%16
// -----------------------------------------------------------------------------
__global__ __launch_bounds__(256)
void proj_split_kernel(const float* __restrict__ x,
                       const float* __restrict__ Wf, const float* __restrict__ bf,
                       const float* __restrict__ Wg, const float* __restrict__ bg,
                       const float* __restrict__ Wh, const float* __restrict__ bh,
                       _Float16* __restrict__ Ghi, _Float16* __restrict__ Glo,
                       _Float16* __restrict__ Fhi, _Float16* __restrict__ Flo,
                       _Float16* __restrict__ Vhi, _Float16* __restrict__ Vlo) {
  int idx = blockIdx.x * blockDim.x + threadIdx.x;   // over B*N*CB = 262144
  int cb = idx & 31;
  int n  = (idx >> 5) & (N_ - 1);
  int b  = idx >> 17;

  const float* xr = x + ((size_t)b * N_ + n) * C_;
  float af = bf[cb], ag = bg[cb], ah = bh[cb];
#pragma unroll 8
  for (int k = 0; k < C_; ++k) {
    float xv = xr[k];
    af += xv * Wf[k * CB_ + cb];
    ag += xv * Wg[k * CB_ + cb];
    ah += xv * Wh[k * CB_ + cb];
  }

  int m  = n & 15;
  int qt = n >> 4;

  // g -> B-fragment layout
  {
    _Float16 hi = (_Float16)ag;
    _Float16 lo = (_Float16)(ag - (float)hi);
    int lane = m + 16 * (((cb & 15) >= 8) ? 1 : 0);
    int slot = (cb & 7) + 8 * (cb >= 16 ? 1 : 0);
    size_t o = ((size_t)(b * QT_ + qt) * 32 + lane) * 16 + slot;
    Ghi[o] = hi; Glo[o] = lo;
  }
  // f -> A-fragment layout (16-key tiles)
  {
    _Float16 hi = (_Float16)af;
    _Float16 lo = (_Float16)(af - (float)hi);
    int hs   = ((cb & 15) >= 8) ? 1 : 0;
    int lane = m + 16 * hs;
    int slot = (cb & 7) + 8 * (cb >= 16 ? 1 : 0);
    size_t o = ((size_t)(b * QT_ + qt) * 32 + lane) * 16 + slot;
    Fhi[o] = hi; Flo[o] = lo;
  }
  // h -> V^T A-fragment layout (32-key tiles, 2 cbar halves)
  {
    _Float16 hi = (_Float16)ah;
    _Float16 lo = (_Float16)(ah - (float)hi);
    int kt   = n >> 5;
    int kk   = n & 31;
    int cbh  = cb >> 4;
    int hs   = (kk >= 16) ? 1 : 0;
    int lane = (cb & 15) + 16 * hs;
    int slot = kk & 15;
    size_t o = (((size_t)(b * KT32_ + kt) * 2 + cbh) * 32 + lane) * 16 + slot;
    Vhi[o] = hi; Vlo[o] = lo;
  }
}

// -----------------------------------------------------------------------------
// Kernel 2: flash attention, transposed formulation. One wave per
// (batch, 16-query tile, K-split). Computes T = S^T = F * G^T via WMMA
// (Ootomo hi/lo 3-term), online softmax per column (query) with one
// permlanex16 per reduction, accumulates O^T = V^T * P^T via WMMA.
// Emits unnormalized partials (m, l, O) per split.
// -----------------------------------------------------------------------------
__global__ __launch_bounds__(128)
void attn_flash_kernel(const _Float16* __restrict__ Ghi, const _Float16* __restrict__ Glo,
                       const _Float16* __restrict__ Fhi, const _Float16* __restrict__ Flo,
                       const _Float16* __restrict__ Vhi, const _Float16* __restrict__ Vlo,
                       float* __restrict__ Opart, float* __restrict__ Mpart,
                       float* __restrict__ Lpart, int NS) {
  const int lane = threadIdx.x & 31;
  const int half = lane >> 4;
  const int wid  = blockIdx.x * (blockDim.x >> 5) + (threadIdx.x >> 5);
  const int s  = wid % NS;
  const int t  = wid / NS;
  const int qt = t % QT_;
  const int b  = t / QT_;

  // Query fragments (B-side), loaded once per wave.
  const v16h gh = *(const v16h*)(Ghi + ((size_t)(b * QT_ + qt) * 32 + lane) * 16);
  const v16h gl = *(const v16h*)(Glo + ((size_t)(b * QT_ + qt) * 32 + lane) * 16);

  float m_run = -INFINITY;
  float l_run = 0.0f;
  v8f O0 = {}; // bhv^T, cbar 0..15  (rows=cb, cols=query)
  v8f O1 = {}; // bhv^T, cbar 16..31

  const int ktPerS = KT32_ / NS;
  const int kt0 = s * ktPerS;

  for (int kt = kt0; kt < kt0 + ktPerS; ++kt) {
    const int t0 = 2 * kt, t1 = 2 * kt + 1;
    // Key fragments (A-side), 16 keys each.
    v16h fa0h = *(const v16h*)(Fhi + ((size_t)(b * QT_ + t0) * 32 + lane) * 16);
    v16h fa0l = *(const v16h*)(Flo + ((size_t)(b * QT_ + t0) * 32 + lane) * 16);
    v16h fa1h = *(const v16h*)(Fhi + ((size_t)(b * QT_ + t1) * 32 + lane) * 16);
    v16h fa1l = *(const v16h*)(Flo + ((size_t)(b * QT_ + t1) * 32 + lane) * 16);
    // Value fragments (A-side of V^T), 32 keys, per cbar half.
    v16h va0h = *(const v16h*)(Vhi + (((size_t)(b * KT32_ + kt) * 2 + 0) * 32 + lane) * 16);
    v16h va0l = *(const v16h*)(Vlo + (((size_t)(b * KT32_ + kt) * 2 + 0) * 32 + lane) * 16);
    v16h va1h = *(const v16h*)(Vhi + (((size_t)(b * KT32_ + kt) * 2 + 1) * 32 + lane) * 16);
    v16h va1l = *(const v16h*)(Vlo + (((size_t)(b * KT32_ + kt) * 2 + 1) * 32 + lane) * 16);

    // T0 = S^T rows(keys) 0..15, T1 = rows 16..31 of this 32-key tile.
    v8f z = {};
    v8f c0 = wmma16x16x32(fa0h, gh, z);
    c0 = wmma16x16x32(fa0h, gl, c0);
    c0 = wmma16x16x32(fa0l, gh, c0);
    v8f c1 = wmma16x16x32(fa1h, gh, z);
    c1 = wmma16x16x32(fa1h, gl, c1);
    c1 = wmma16x16x32(fa1l, gh, c1);

    // Per-query (column) max over the 32 keys of this tile.
    float mt = c0[0];
#pragma unroll
    for (int i = 1; i < 8; ++i) mt = fmaxf(mt, c0[i]);
#pragma unroll
    for (int i = 0; i < 8; ++i) mt = fmaxf(mt, c1[i]);
    mt = fmaxf(mt, xor16(mt));

    const float mnew  = fmaxf(m_run, mt);
    const float alpha = __expf(m_run - mnew);

    float p0[8], p1[8];
    float rs = 0.0f;
#pragma unroll
    for (int i = 0; i < 8; ++i) { p0[i] = __expf(c0[i] - mnew); rs += p0[i]; }
#pragma unroll
    for (int i = 0; i < 8; ++i) { p1[i] = __expf(c1[i] - mnew); rs += p1[i]; }
    rs += xor16(rs);

    l_run = l_run * alpha + rs;
    m_run = mnew;
#pragma unroll
    for (int i = 0; i < 8; ++i) { O0[i] *= alpha; O1[i] *= alpha; }

    // Build P^T as a B-fragment (K=32 keys, N=16 queries). Half-0 lanes only
    // need the partner's p0[i]; half-1 lanes only need the partner's p1[i].
    // Pre-select, then a single permlanex16 per register does the exchange.
    v16h pbh = {}, pbl = {};
#pragma unroll
    for (int i = 0; i < 8; ++i) {
      float send  = half ? p0[i] : p1[i];
      float cross = xor16(send);          // half0: partner p0; half1: partner p1
      float vA = half ? cross : p0[i];    // slots 0..7 : keys 0..7 (h0) / 16..23 (h1)
      float vB = half ? p1[i] : cross;    // slots 8..15: keys 8..15 (h0) / 24..31 (h1)
      _Float16 hA = (_Float16)vA;
      _Float16 hB = (_Float16)vB;
      pbh[i]     = hA; pbl[i]     = (_Float16)(vA - (float)hA);
      pbh[i + 8] = hB; pbl[i + 8] = (_Float16)(vB - (float)hB);
    }

    O0 = wmma16x16x32(va0h, pbh, O0);
    O0 = wmma16x16x32(va0h, pbl, O0);
    O0 = wmma16x16x32(va0l, pbh, O0);
    O1 = wmma16x16x32(va1h, pbh, O1);
    O1 = wmma16x16x32(va1h, pbl, O1);
    O1 = wmma16x16x32(va1l, pbh, O1);
  }

  // Emit partials: [b][s][q][cb]
  const int q = qt * 16 + (lane & 15);
  const size_t qbase = (size_t)(b * NS + s) * N_ + q;
  if (half == 0) { Mpart[qbase] = m_run; Lpart[qbase] = l_run; }
#pragma unroll
  for (int i = 0; i < 8; ++i) {
    int cb = i + 8 * half;
    Opart[qbase * CB_ + cb]      = O0[i];
    Opart[qbase * CB_ + cb + 16] = O1[i];
  }
}

// -----------------------------------------------------------------------------
// Kernel 3: combine K-splits -> normalized bhv[b][q][cb].
// -----------------------------------------------------------------------------
__global__ __launch_bounds__(256)
void combine_kernel(const float* __restrict__ Opart, const float* __restrict__ Mpart,
                    const float* __restrict__ Lpart, float* __restrict__ bhv, int NS) {
  int idx = blockIdx.x * blockDim.x + threadIdx.x;  // over B*N
  if (idx >= B_ * N_) return;
  int q = idx & (N_ - 1);
  int b = idx >> 12;

  float M = -INFINITY;
  for (int s = 0; s < NS; ++s)
    M = fmaxf(M, Mpart[(size_t)(b * NS + s) * N_ + q]);

  float L = 0.0f;
  float acc[CB_];
#pragma unroll
  for (int cb = 0; cb < CB_; ++cb) acc[cb] = 0.0f;

  for (int s = 0; s < NS; ++s) {
    size_t o = (size_t)(b * NS + s) * N_ + q;
    float w = __expf(Mpart[o] - M);
    L += w * Lpart[o];
    const float* op = Opart + o * CB_;
#pragma unroll
    for (int cb = 0; cb < CB_; ++cb) acc[cb] += w * op[cb];
  }
  float inv = 1.0f / L;
  float* dst = bhv + ((size_t)b * N_ + q) * CB_;
#pragma unroll
  for (int cb = 0; cb < CB_; ++cb) dst[cb] = acc[cb] * inv;
}

// -----------------------------------------------------------------------------
// Kernel 4: epilogue, out = x + gamma * (bhv @ Wo + bo), exact fp32.
// -----------------------------------------------------------------------------
__global__ __launch_bounds__(256)
void epilogue_kernel(const float* __restrict__ x, const float* __restrict__ bhv,
                     const float* __restrict__ Wo, const float* __restrict__ bo,
                     const float* __restrict__ gamma, float* __restrict__ out) {
  int idx = blockIdx.x * blockDim.x + threadIdx.x;  // over B*N*C
  int c = idx & (C_ - 1);
  int n = (idx >> 6) & (N_ - 1);
  int b = idx >> 18;
  const float* br = bhv + ((size_t)b * N_ + n) * CB_;
  float acc = bo[c];
#pragma unroll 8
  for (int cb = 0; cb < CB_; ++cb) acc += br[cb] * Wo[cb * C_ + c];
  out[idx] = x[idx] + gamma[0] * acc;
}

// -----------------------------------------------------------------------------
extern "C" void kernel_launch(void* const* d_in, const int* in_sizes, int n_in,
                              void* d_out, int out_size, void* d_ws, size_t ws_size,
                              hipStream_t stream) {
  (void)in_sizes; (void)n_in; (void)out_size;
  const float* x     = (const float*)d_in[0];
  const float* Wf    = (const float*)d_in[1];
  const float* bf    = (const float*)d_in[2];
  const float* Wg    = (const float*)d_in[3];
  const float* bg    = (const float*)d_in[4];
  const float* Wh    = (const float*)d_in[5];
  const float* bh    = (const float*)d_in[6];
  const float* Wo    = (const float*)d_in[7];
  const float* bo    = (const float*)d_in[8];
  const float* gamma = (const float*)d_in[9];
  float* out = (float*)d_out;

  char* base = (char*)d_ws;
  size_t off = 0;
  auto take = [&](size_t bytes) -> char* {
    char* p = base + off;
    off = (off + bytes + 255) & ~(size_t)255;
    return p;
  };

  const size_t halfBuf = (size_t)B_ * N_ * CB_ * sizeof(_Float16);  // 512 KB each
  _Float16* Ghi = (_Float16*)take(halfBuf);
  _Float16* Glo = (_Float16*)take(halfBuf);
  _Float16* Fhi = (_Float16*)take(halfBuf);
  _Float16* Flo = (_Float16*)take(halfBuf);
  _Float16* Vhi = (_Float16*)take(halfBuf);
  _Float16* Vlo = (_Float16*)take(halfBuf);
  float* bhv = (float*)take((size_t)B_ * N_ * CB_ * sizeof(float));  // 1 MB

  // Choose K-split count from available workspace (deterministic).
  const size_t perSplit = (size_t)B_ * N_ * CB_ * 4 + 2 * (size_t)B_ * N_ * 4 + 1024;
  int NS = 1;
  if (off + 8 * perSplit <= ws_size) NS = 8;
  else if (off + 4 * perSplit <= ws_size) NS = 4;
  else if (off + 2 * perSplit <= ws_size) NS = 2;

  float* Opart = (float*)take((size_t)NS * B_ * N_ * CB_ * sizeof(float));
  float* Mpart = (float*)take((size_t)NS * B_ * N_ * sizeof(float));
  float* Lpart = (float*)take((size_t)NS * B_ * N_ * sizeof(float));

  // 1) projections + fragment pre-swizzle
  {
    int total = B_ * N_ * CB_;  // 262144
    proj_split_kernel<<<total / 256, 256, 0, stream>>>(
        x, Wf, bf, Wg, bg, Wh, bh, Ghi, Glo, Fhi, Flo, Vhi, Vlo);
  }
  // 2) flash attention (one wave per (b, qtile, split))
  {
    int units = B_ * QT_ * NS;          // waves
    attn_flash_kernel<<<units / 4, 128, 0, stream>>>(
        Ghi, Glo, Fhi, Flo, Vhi, Vlo, Opart, Mpart, Lpart, NS);
  }
  // 3) combine splits
  {
    int total = B_ * N_;  // 8192
    combine_kernel<<<total / 256, 256, 0, stream>>>(Opart, Mpart, Lpart, bhv, NS);
  }
  // 4) epilogue
  {
    int total = B_ * N_ * C_;  // 524288
    epilogue_kernel<<<total / 256, 256, 0, stream>>>(x, bhv, Wo, bo, gamma, out);
  }
}
